// DualEdgeConv_42236708389565
// MI455X (gfx1250) — matrix operand order repaired and checked by reading, not compiled, not run on hardware
//
#include <hip/hip_runtime.h>
#include <hip/hip_bf16.h>

typedef __attribute__((ext_vector_type(16))) __bf16 v16bf;
typedef __attribute__((ext_vector_type(8)))  __bf16 v8bf;
typedef __attribute__((ext_vector_type(8)))  float  v8f;

#define N_NODES 50000
#define N_EDGES 800000
#define DIM 64
#define HID 128
#define TILES_PER_GRAPH (N_EDGES / 16)
#define TILES_TOTAL (2 * TILES_PER_GRAPH)
#define STG 136               // padded LDS row stride (bf16 elems): 272B = 68 banks -> conflict-free
#define WAVES 8
#define THREADS (WAVES * 32)

union V16 { v16bf v; v8bf h[2]; };

__global__ void dec_init_kernel(float* out, int n) {
#if defined(__HIP_DEVICE_COMPILE__)
  int i = blockIdx.x * blockDim.x + threadIdx.x;
  int stride = gridDim.x * blockDim.x;
  for (; i < n; i += stride) out[i] = -INFINITY;
#endif
}

__global__ void dec_finalize_kernel(float* out, int n) {
#if defined(__HIP_DEVICE_COMPILE__)
  int i = blockIdx.x * blockDim.x + threadIdx.x;
  int stride = gridDim.x * blockDim.x;
  for (; i < n; i += stride) {
    float v = out[i];
    out[i] = (v >= -3.3e38f) ? v : 0.0f;   // -inf (empty segment) and NaN -> 0
  }
#endif
}

// Non-returning device-scope float atomic max (CDNA5 GLOBAL_ATOMIC_MAX_NUM_F32,
// tracked with STOREcnt; no pre-op value returned).
__device__ __forceinline__ void atomic_max_f32(float* p, float v) {
  asm volatile("global_atomic_max_num_f32 %0, %1, off scope:SCOPE_DEV"
               :: "v"(p), "v"(v) : "memory");
}

__global__ __launch_bounds__(THREADS)
void dec_edgeconv_kernel(const float* __restrict__ x1, const float* __restrict__ x2,
                         const int* __restrict__ ei1, const int* __restrict__ ei2,
                         const float* __restrict__ W1, const float* __restrict__ b1,
                         const float* __restrict__ W2, const float* __restrict__ b2,
                         float* __restrict__ out) {
#if defined(__HIP_DEVICE_COMPILE__)
  extern __shared__ __align__(16) char smem[];
  __bf16* W1T      = (__bf16*)smem;            // [128][STG]  W1T[n][k] = W1[k][n]
  __bf16* W2T      = W1T + 128 * STG;          // [ 64][STG]  W2T[n][k] = W2[k][n]
  __bf16* stageAll = W2T + 64 * STG;           // WAVES x [16][STG]

  const int lane   = threadIdx.x & 31;
  const int wave   = threadIdx.x >> 5;
  const int lane16 = lane & 15;
  const int laneHi = lane >> 4;

  // --- one-time per block: weights -> bf16, transposed, padded ---
  for (int idx = threadIdx.x; idx < 128 * 128; idx += THREADS) {
    int k = idx >> 7, n = idx & 127;
    W1T[n * STG + k] = (__bf16)W1[idx];
  }
  for (int idx = threadIdx.x; idx < 128 * 64; idx += THREADS) {
    int k = idx >> 6, n = idx & 63;
    W2T[n * STG + k] = (__bf16)W2[idx];
  }
  __syncthreads();

  __bf16* stage = stageAll + wave * 16 * STG;

  for (int tile = blockIdx.x * WAVES + wave; tile < TILES_TOTAL; tile += gridDim.x * WAVES) {
    const int g     = (tile >= TILES_PER_GRAPH) ? 1 : 0;
    const int t     = tile - g * TILES_PER_GRAPH;
    const int ebase = t * 16;
    const int* eRow = g ? ei2 : ei1;     // edge_index[0]
    const int* eCol = eRow + N_EDGES;    // edge_index[1]
    const float* xd = g ? x2 : x1;

    // --- gather edge features into stage: row m = [x1[r] | xd[c]-x1[r]] (bf16) ---
    {
      const int m = lane16;
      const int r = eRow[ebase + m];
      __bf16* srow = stage + m * STG;
      if (laneHi == 0) {
        const float4* ps = (const float4*)(x1 + (size_t)r * DIM);
        #pragma unroll
        for (int i = 0; i < 16; ++i) {
          float4 a = ps[i];
          srow[4 * i + 0] = (__bf16)a.x; srow[4 * i + 1] = (__bf16)a.y;
          srow[4 * i + 2] = (__bf16)a.z; srow[4 * i + 3] = (__bf16)a.w;
        }
      } else {
        const int c = eCol[ebase + m];
        const float4* ps = (const float4*)(x1 + (size_t)r * DIM);
        const float4* pd = (const float4*)(xd + (size_t)c * DIM);
        #pragma unroll
        for (int i = 0; i < 16; ++i) {
          float4 a = ps[i]; float4 d = pd[i];
          srow[64 + 4 * i + 0] = (__bf16)(d.x - a.x); srow[64 + 4 * i + 1] = (__bf16)(d.y - a.y);
          srow[64 + 4 * i + 2] = (__bf16)(d.z - a.z); srow[64 + 4 * i + 3] = (__bf16)(d.w - a.w);
        }
      }
    }

    // --- layer 1: h[16x128] = relu(e[16x128] @ W1 + b1), via 8 N-tiles x 4 K-steps ---
    V16 a1[4];
    #pragma unroll
    for (int kb = 0; kb < 4; ++kb) {
      // A 16x32 bf16 layout: lanes 0-15 K {0..7,16..23}, lanes 16-31 K {8..15,24..31}
      const __bf16* p = stage + lane16 * STG + kb * 32 + laneHi * 8;
      a1[kb].h[0] = *(const v8bf*)p;
      a1[kb].h[1] = *(const v8bf*)(p + 16);
    }
    #pragma unroll
    for (int nb = 0; nb < 8; ++nb) {
      v8f acc = {};
      #pragma unroll
      for (int kb = 0; kb < 4; ++kb) {
        V16 b;  // B 32x16: lane = col, lanes 0-15 K {0..7,16..23}, lanes 16-31 K {8..15,24..31}
        const __bf16* p = W1T + (nb * 16 + lane16) * STG + kb * 32 + laneHi * 8;
        b.h[0] = *(const v8bf*)p;
        b.h[1] = *(const v8bf*)(p + 16);
        acc = __builtin_amdgcn_wmma_f32_16x16x32_bf16(false, a1[kb].v, false, b.v,
                                                      (short)0, acc, false, false);
      }
      const float bias = b1[nb * 16 + lane16];
      #pragma unroll
      for (int r = 0; r < 8; ++r) {
        float hv = acc[r] + bias;
        hv = hv > 0.0f ? hv : 0.0f;
        // C/D layout: VGPR r -> M = r + 8*laneHi, lane16 -> N
        stage[(r + laneHi * 8) * STG + nb * 16 + lane16] = (__bf16)hv;
      }
    }

    // --- layer 2: o[16x64] = h @ W2 + b2, then scatter-max by source row ---
    V16 a2[4];
    #pragma unroll
    for (int kb = 0; kb < 4; ++kb) {
      const __bf16* p = stage + lane16 * STG + kb * 32 + laneHi * 8;
      a2[kb].h[0] = *(const v8bf*)p;
      a2[kb].h[1] = *(const v8bf*)(p + 16);
    }
    int rows8[8];
    #pragma unroll
    for (int r = 0; r < 8; ++r) rows8[r] = eRow[ebase + r + laneHi * 8];

    #pragma unroll
    for (int nb = 0; nb < 4; ++nb) {
      v8f acc = {};
      #pragma unroll
      for (int kb = 0; kb < 4; ++kb) {
        V16 b;
        const __bf16* p = W2T + (nb * 16 + lane16) * STG + kb * 32 + laneHi * 8;
        b.h[0] = *(const v8bf*)p;
        b.h[1] = *(const v8bf*)(p + 16);
        acc = __builtin_amdgcn_wmma_f32_16x16x32_bf16(false, a2[kb].v, false, b.v,
                                                      (short)0, acc, false, false);
      }
      const float bias = b2[nb * 16 + lane16];
      const int col = g * 64 + nb * 16 + lane16;
      #pragma unroll
      for (int r = 0; r < 8; ++r) {
        float v = acc[r] + bias;
        // lanes 0-15 cover 16 consecutive floats of one row -> 1 cacheline per half-wave
        atomic_max_f32(out + (size_t)rows8[r] * (2 * DIM) + col, v);
      }
    }
  }
#endif
}

extern "C" void kernel_launch(void* const* d_in, const int* in_sizes, int n_in,
                              void* d_out, int out_size, void* d_ws, size_t ws_size,
                              hipStream_t stream) {
  (void)in_sizes; (void)n_in; (void)d_ws; (void)ws_size;
  const float* x1  = (const float*)d_in[0];
  const float* x2  = (const float*)d_in[1];
  const int*   ei1 = (const int*)d_in[2];
  const int*   ei2 = (const int*)d_in[3];
  const float* W1  = (const float*)d_in[4];
  const float* b1  = (const float*)d_in[5];
  const float* W2  = (const float*)d_in[6];
  const float* b2  = (const float*)d_in[7];
  float* out = (float*)d_out;

  const int total = out_size;  // N * 2D = 6,400,000 floats

  dec_init_kernel<<<4096, 256, 0, stream>>>(out, total);

  const size_t shmem = (size_t)(128 * STG + 64 * STG + WAVES * 16 * STG) * 2 /*bytes per bf16*/;
  dec_edgeconv_kernel<<<1024, THREADS, shmem, stream>>>(x1, x2, ei1, ei2, W1, b1, W2, b2, out);

  dec_finalize_kernel<<<4096, 256, 0, stream>>>(out, total);
}
